// ErnieBlock_12850542150234
// MI455X (gfx1250) — matrix operand order repaired and checked by reading, not compiled, not run on hardware
//
#include <hip/hip_runtime.h>
#include <math.h>

#define S_LEN 2048
#define DIM   1024
#define HEADS 16
#define HDIM  64
#define NEXP  8
#define DFF   2048
#define LDK   40          // padded LDS row stride (bf16 elements) for a 32-wide K tile
#define TILE_ELEMS (128 * LDK)

typedef __attribute__((ext_vector_type(16))) __bf16 v16bf;
typedef __attribute__((ext_vector_type(8)))  __bf16 v8bf;
typedef __attribute__((ext_vector_type(8)))  float  v8f;

union V16U { v16bf v; v8bf h[2]; };

__device__ __forceinline__ __bf16 f2bf(float f) {
  union { float f; unsigned u; } x; x.f = f;
  unsigned r = x.u + 0x7fffu + ((x.u >> 16) & 1u);   // round-to-nearest-even
  unsigned short hs = (unsigned short)(r >> 16);
  __bf16 b; __builtin_memcpy(&b, &hs, 2); return b;
}

__device__ __forceinline__ float gelu_tanh(float x) {
  float x3 = x * x * x;
  return 0.5f * x * (1.0f + tanhf(0.7978845608028654f * (x + 0.044715f * x3)));
}

// ---------------------------------------------------------------------------
// Stage one 128x32 A tile (bf16 activations) and one 128x32 B tile
// (f32 weights, converted to bf16) into LDS, layout [row(m|n)][k], stride LDK.
// BT=true : W is f32 [N,K] (torch weight, computing A @ W^T)
// BT=false: W is f32 [K,N] (w1/w2 layout, transposed into LDS)
// Also issues a one-step-ahead global_prefetch of the next chunk into GL2.
// ---------------------------------------------------------------------------
template <bool BT>
__device__ __forceinline__ void stage_tiles(const __bf16* __restrict__ A,
                                            const float* __restrict__ W,
                                            int N, int K, int m0, int n0, int k0,
                                            __bf16* As, __bf16* Bs) {
  const int t = threadIdx.x;
  {
    int mr = t >> 1, kh = (t & 1) * 16;
    const __bf16* gp = A + (size_t)(m0 + mr) * K + k0 + kh;
    if (k0 + 32 < K) __builtin_prefetch(gp + 32, 0, 1);
    const v8bf* src = (const v8bf*)gp;
    v8bf a0 = src[0], a1 = src[1];
    *(v8bf*)(As + mr * LDK + kh)     = a0;
    *(v8bf*)(As + mr * LDK + kh + 8) = a1;
  }
  if (BT) { // W[N,K]: B[k,n] = W[n,k]; store Bs[n][k] (contiguous along k)
    int nr = t >> 1, kh = (t & 1) * 16;
    const float* gp = W + (size_t)(n0 + nr) * K + k0 + kh;
    if (k0 + 32 < K) __builtin_prefetch(gp + 32, 0, 1);
    const float4* src = (const float4*)gp;
    float4 f0 = src[0], f1 = src[1], f2 = src[2], f3 = src[3];
    v8bf lo, hi;
    lo[0]=f2bf(f0.x); lo[1]=f2bf(f0.y); lo[2]=f2bf(f0.z); lo[3]=f2bf(f0.w);
    lo[4]=f2bf(f1.x); lo[5]=f2bf(f1.y); lo[6]=f2bf(f1.z); lo[7]=f2bf(f1.w);
    hi[0]=f2bf(f2.x); hi[1]=f2bf(f2.y); hi[2]=f2bf(f2.z); hi[3]=f2bf(f2.w);
    hi[4]=f2bf(f3.x); hi[5]=f2bf(f3.y); hi[6]=f2bf(f3.z); hi[7]=f2bf(f3.w);
    *(v8bf*)(Bs + nr * LDK + kh)     = lo;
    *(v8bf*)(Bs + nr * LDK + kh + 8) = hi;
  } else { // W[K,N]: transpose into Bs[n][k]
    int kr = t >> 3, nc = (t & 7) * 16;
    const float* gp = W + (size_t)(k0 + kr) * N + n0 + nc;
    if (k0 + 32 < K) __builtin_prefetch(gp + (size_t)32 * N, 0, 1);
    const float4* src = (const float4*)gp;
#pragma unroll
    for (int j4 = 0; j4 < 4; ++j4) {
      float4 f = src[j4];
      Bs[(nc + j4 * 4 + 0) * LDK + kr] = f2bf(f.x);
      Bs[(nc + j4 * 4 + 1) * LDK + kr] = f2bf(f.y);
      Bs[(nc + j4 * 4 + 2) * LDK + kr] = f2bf(f.z);
      Bs[(nc + j4 * 4 + 3) * LDK + kr] = f2bf(f.w);
    }
  }
}

// ---------------------------------------------------------------------------
// Core WMMA GEMM, double-buffered LDS: C[128x128] block tile, 8 waves (2x4),
// each wave 64x32 = eight 16x16 f32 accumulators. One barrier per K-step:
// staging of step k+1 overlaps the 8 WMMAs of step k.
// ---------------------------------------------------------------------------
template <bool BT>
__device__ __forceinline__ void gemm_main(const __bf16* __restrict__ A,
                                          const float* __restrict__ W,
                                          int N, int K, int m0, int n0,
                                          __bf16* As, __bf16* Bs, v8f acc[8]) {
  const int t    = threadIdx.x;
  const int wave = t >> 5, lane = t & 31;
  const int lh   = lane >> 4, l15 = lane & 15;
  const int m0w  = (wave >> 2) * 64;
  const int n0w  = (wave & 3) * 32;

  v8f z = {};
#pragma unroll
  for (int i = 0; i < 8; ++i) acc[i] = z;

  stage_tiles<BT>(A, W, N, K, m0, n0, 0, As, Bs);

  for (int k0 = 0; k0 < K; k0 += 32) {
    const int buf = (k0 >> 5) & 1;
    __bf16* Ab = As + buf * TILE_ELEMS;
    __bf16* Bb = Bs + buf * TILE_ELEMS;
    __syncthreads();
    if (k0 + 32 < K)
      stage_tiles<BT>(A, W, N, K, m0, n0, k0 + 32,
                      As + (buf ^ 1) * TILE_ELEMS, Bs + (buf ^ 1) * TILE_ELEMS);

    V16U a[4], b[2];
#pragma unroll
    for (int mt = 0; mt < 4; ++mt) {
      const __bf16* ap = Ab + (m0w + mt * 16 + l15) * LDK + lh * 8;
      a[mt].h[0] = *(const v8bf*)ap;          // K = lh*8 .. +7
      a[mt].h[1] = *(const v8bf*)(ap + 16);   // K = 16+lh*8 .. +7
    }
#pragma unroll
    for (int nt = 0; nt < 2; ++nt) {
      const __bf16* bp = Bb + (n0w + nt * 16 + l15) * LDK + lh * 8;
      b[nt].h[0] = *(const v8bf*)bp;          // K = lh*8 .. +7
      b[nt].h[1] = *(const v8bf*)(bp + 16);   // K = 16+lh*8 .. +7
    }
#pragma unroll
    for (int mt = 0; mt < 4; ++mt)
#pragma unroll
      for (int nt = 0; nt < 2; ++nt)
        acc[mt * 2 + nt] = __builtin_amdgcn_wmma_f32_16x16x32_bf16(
            false, a[mt].v, false, b[nt].v, (short)0, acc[mt * 2 + nt], false, false);
  }
}

// C/D layout: element i of v8f -> row = i + 8*(lane>>4), col = lane&15.
#define EPI_BEGIN                                                         \
  const int t_ = threadIdx.x;                                             \
  const int wave_ = t_ >> 5, lane_ = t_ & 31;                             \
  const int lh_ = lane_ >> 4, l15_ = lane_ & 15;                          \
  const int m0w_ = (wave_ >> 2) * 64, n0w_ = (wave_ & 3) * 32;            \
  _Pragma("unroll") for (int mt = 0; mt < 4; ++mt)                        \
  _Pragma("unroll") for (int nt = 0; nt < 2; ++nt)                        \
  _Pragma("unroll") for (int i = 0; i < 8; ++i) {                         \
    const int m = m0 + m0w_ + mt * 16 + i + 8 * lh_;                      \
    const int n = n0 + n0w_ + nt * 16 + l15_;                             \
    const float v = acc[mt * 2 + nt][i];
#define EPI_END }

// ---------------------------------------------------------------------------
__global__ void __launch_bounds__(256) k_ln1(const float* __restrict__ x,
                                             const float* __restrict__ g,
                                             const float* __restrict__ b,
                                             __bf16* __restrict__ h) {
  __shared__ float red[256];
  const int row = blockIdx.x, t = threadIdx.x;
  const float* xr = x + (size_t)row * DIM;
  float v[4];
#pragma unroll
  for (int j = 0; j < 4; ++j) v[j] = xr[t * 4 + j];
  red[t] = v[0] + v[1] + v[2] + v[3];
  __syncthreads();
  for (int st = 128; st > 0; st >>= 1) { if (t < st) red[t] += red[t + st]; __syncthreads(); }
  float mu = red[0] * (1.0f / DIM);
  __syncthreads();
  float vs = 0.f;
#pragma unroll
  for (int j = 0; j < 4; ++j) { float d = v[j] - mu; vs += d * d; }
  red[t] = vs;
  __syncthreads();
  for (int st = 128; st > 0; st >>= 1) { if (t < st) red[t] += red[t + st]; __syncthreads(); }
  float rstd = rsqrtf(red[0] * (1.0f / DIM) + 1e-5f);
#pragma unroll
  for (int j = 0; j < 4; ++j) {
    int c = t * 4 + j;
    h[(size_t)row * DIM + c] = f2bf((v[j] - mu) * rstd * g[c] + b[c]);
  }
}

// QKV: qkv = h @ in_proj_w^T + b; Q,K as [H,S,64] bf16, V transposed [H,64,S].
// Each 128-wide n block lies entirely in one of the Q/K/V sections -> the
// section test is block-uniform (single scalar branch, straight-line stores).
__global__ void __launch_bounds__(256) k_gemm_qkv(const __bf16* __restrict__ A,
                                                  const float* __restrict__ W,
                                                  const float* __restrict__ bias,
                                                  __bf16* __restrict__ Qb,
                                                  __bf16* __restrict__ Kb,
                                                  __bf16* __restrict__ Vt) {
  __shared__ __bf16 As[2 * TILE_ELEMS], Bs[2 * TILE_ELEMS];
  const int n0 = blockIdx.x * 128, m0 = blockIdx.y * 128;
  v8f acc[8];
  gemm_main<true>(A, W, 3 * DIM, DIM, m0, n0, As, Bs, acc);
  const int sec = n0 >> 10;                 // 0=Q, 1=K, 2=V (uniform per block)
  if (sec == 0) {
    EPI_BEGIN
      int nn = n & 1023, hh = nn >> 6, dd = nn & 63;
      Qb[((size_t)hh * S_LEN + m) * HDIM + dd] = f2bf(v + bias[n]);
    EPI_END
  } else if (sec == 1) {
    EPI_BEGIN
      int nn = n & 1023, hh = nn >> 6, dd = nn & 63;
      Kb[((size_t)hh * S_LEN + m) * HDIM + dd] = f2bf(v + bias[n]);
    EPI_END
  } else {
    EPI_BEGIN
      int nn = n & 1023, hh = nn >> 6, dd = nn & 63;
      Vt[((size_t)hh * HDIM + dd) * S_LEN + m] = f2bf(v + bias[n]);
    EPI_END
  }
}

// Flash attention: 1 wave per 16-query tile per head; 4 waves per block.
__global__ void __launch_bounds__(128) k_attn(const __bf16* __restrict__ Q,
                                              const __bf16* __restrict__ K,
                                              const __bf16* __restrict__ Vt,
                                              __bf16* __restrict__ ctx) {
  __shared__ __bf16 P[4][16 * LDK];
  const int t = threadIdx.x, wave = t >> 5, lane = t & 31;
  const int lh = lane >> 4, l15 = lane & 15;
  const int hh = blockIdx.y;
  const int q0 = blockIdx.x * 64 + wave * 16;

  const __bf16* qrow = Q + ((size_t)hh * S_LEN + q0 + l15) * HDIM;
  V16U aq0, aq1;
  aq0.h[0] = *(const v8bf*)(qrow + lh * 16);
  aq0.h[1] = *(const v8bf*)(qrow + lh * 16 + 8);
  aq1.h[0] = *(const v8bf*)(qrow + 32 + lh * 16);
  aq1.h[1] = *(const v8bf*)(qrow + 32 + lh * 16 + 8);

  v8f z = {};
  v8f od[4] = {z, z, z, z};
  float rmax[8], rsum[8];
#pragma unroll
  for (int i = 0; i < 8; ++i) { rmax[i] = -3.0e38f; rsum[i] = 0.f; }

  for (int kb = 0; kb < S_LEN; kb += 32) {
    v8f s0 = z, s1 = z;
    {
      const __bf16* kr0 = K + ((size_t)hh * S_LEN + kb + l15) * HDIM;
      const __bf16* kr1 = K + ((size_t)hh * S_LEN + kb + 16 + l15) * HDIM;
      V16U b00, b01, b10, b11;
      b00.h[0] = *(const v8bf*)(kr0 + lh * 16);      b00.h[1] = *(const v8bf*)(kr0 + lh * 16 + 8);
      b01.h[0] = *(const v8bf*)(kr0 + 32 + lh * 16); b01.h[1] = *(const v8bf*)(kr0 + 32 + lh * 16 + 8);
      b10.h[0] = *(const v8bf*)(kr1 + lh * 16);      b10.h[1] = *(const v8bf*)(kr1 + lh * 16 + 8);
      b11.h[0] = *(const v8bf*)(kr1 + 32 + lh * 16); b11.h[1] = *(const v8bf*)(kr1 + 32 + lh * 16 + 8);
      s0 = __builtin_amdgcn_wmma_f32_16x16x32_bf16(false, aq0.v, false, b00.v, (short)0, s0, false, false);
      s0 = __builtin_amdgcn_wmma_f32_16x16x32_bf16(false, aq1.v, false, b01.v, (short)0, s0, false, false);
      s1 = __builtin_amdgcn_wmma_f32_16x16x32_bf16(false, aq0.v, false, b10.v, (short)0, s1, false, false);
      s1 = __builtin_amdgcn_wmma_f32_16x16x32_bf16(false, aq1.v, false, b11.v, (short)0, s1, false, false);
    }
    float tmax[8];
#pragma unroll
    for (int i = 0; i < 8; ++i) {
      float a0 = s0[i] * 0.125f, a1 = s1[i] * 0.125f;   // 1/sqrt(64)
      s0[i] = a0; s1[i] = a1;
      tmax[i] = fmaxf(a0, a1);
    }
#pragma unroll
    for (int i = 0; i < 8; ++i)
      for (int off = 1; off < 16; off <<= 1) tmax[i] = fmaxf(tmax[i], __shfl_xor(tmax[i], off));
    float fac[8];
#pragma unroll
    for (int i = 0; i < 8; ++i) {
      float mn = fmaxf(rmax[i], tmax[i]);
      fac[i] = __expf(rmax[i] - mn);
      rmax[i] = mn;
    }
    float p0[8], p1[8], ps[8];
#pragma unroll
    for (int i = 0; i < 8; ++i) {
      p0[i] = __expf(s0[i] - rmax[i]);
      p1[i] = __expf(s1[i] - rmax[i]);
      ps[i] = p0[i] + p1[i];
    }
#pragma unroll
    for (int i = 0; i < 8; ++i)
      for (int off = 1; off < 16; off <<= 1) ps[i] += __shfl_xor(ps[i], off);
#pragma unroll
    for (int i = 0; i < 8; ++i) rsum[i] = rsum[i] * fac[i] + ps[i];
#pragma unroll
    for (int j = 0; j < 4; ++j)
#pragma unroll
      for (int i = 0; i < 8; ++i) od[j][i] = od[j][i] * fac[i];

    // relayout P: C-layout (row in element, col in lane) -> A-layout via LDS
    __syncthreads();
    __bf16* pw = &P[wave][0];
#pragma unroll
    for (int i = 0; i < 8; ++i) {
      int r = i + 8 * lh;
      pw[r * LDK + l15]      = f2bf(p0[i]);
      pw[r * LDK + 16 + l15] = f2bf(p1[i]);
    }
    __syncthreads();
    V16U ap;
    const __bf16* prd = pw + l15 * LDK + lh * 8;
    ap.h[0] = *(const v8bf*)prd;
    ap.h[1] = *(const v8bf*)(prd + 16);
#pragma unroll
    for (int j = 0; j < 4; ++j) {
      const __bf16* vr = Vt + ((size_t)hh * HDIM + j * 16 + l15) * S_LEN + kb + lh * 8;
      V16U bv; bv.h[0] = *(const v8bf*)vr; bv.h[1] = *(const v8bf*)(vr + 16);
      od[j] = __builtin_amdgcn_wmma_f32_16x16x32_bf16(false, ap.v, false, bv.v, (short)0, od[j], false, false);
    }
  }
  float inv[8];
#pragma unroll
  for (int i = 0; i < 8; ++i) inv[i] = (rsum[i] > 0.f) ? 1.f / rsum[i] : 0.f;
#pragma unroll
  for (int j = 0; j < 4; ++j)
#pragma unroll
    for (int i = 0; i < 8; ++i)
      ctx[(size_t)(q0 + i + 8 * lh) * DIM + hh * HDIM + j * 16 + l15] = f2bf(od[j][i] * inv[i]);
}

// out-proj + residual: x1 = x + ctx @ out_w^T + b  -> d_out (f32)
__global__ void __launch_bounds__(256) k_gemm_out(const __bf16* __restrict__ A,
                                                  const float* __restrict__ W,
                                                  const float* __restrict__ bias,
                                                  const float* __restrict__ xres,
                                                  float* __restrict__ out) {
  __shared__ __bf16 As[2 * TILE_ELEMS], Bs[2 * TILE_ELEMS];
  const int n0 = blockIdx.x * 128, m0 = blockIdx.y * 128;
  v8f acc[8];
  gemm_main<true>(A, W, DIM, DIM, m0, n0, As, Bs, acc);
  EPI_BEGIN
    out[(size_t)m * DIM + n] = v + bias[n] + xres[(size_t)m * DIM + n];
  EPI_END
}

// LN2 + router logits + top-2 combine weights
__global__ void __launch_bounds__(256) k_ln2_router(const float* __restrict__ x1,
                                                    const float* __restrict__ g,
                                                    const float* __restrict__ b,
                                                    const float* __restrict__ gw,
                                                    const float* __restrict__ gb,
                                                    __bf16* __restrict__ h2,
                                                    float* __restrict__ logits,
                                                    float* __restrict__ cw) {
  __shared__ float red[256];
  __shared__ float norm[DIM];
  __shared__ float lg[NEXP];
  const int row = blockIdx.x, t = threadIdx.x;
  const float* xr = x1 + (size_t)row * DIM;
  float v[4];
#pragma unroll
  for (int j = 0; j < 4; ++j) v[j] = xr[t * 4 + j];
  red[t] = v[0] + v[1] + v[2] + v[3];
  __syncthreads();
  for (int st = 128; st > 0; st >>= 1) { if (t < st) red[t] += red[t + st]; __syncthreads(); }
  float mu = red[0] * (1.0f / DIM);
  __syncthreads();
  float vs = 0.f;
#pragma unroll
  for (int j = 0; j < 4; ++j) { float d = v[j] - mu; vs += d * d; }
  red[t] = vs;
  __syncthreads();
  for (int st = 128; st > 0; st >>= 1) { if (t < st) red[t] += red[t + st]; __syncthreads(); }
  float rstd = rsqrtf(red[0] * (1.0f / DIM) + 1e-5f);
  __syncthreads();
#pragma unroll
  for (int j = 0; j < 4; ++j) {
    int c = t * 4 + j;
    float nv = (v[j] - mu) * rstd * g[c] + b[c];
    norm[c] = nv;
    h2[(size_t)row * DIM + c] = f2bf(nv);
  }
  __syncthreads();
  // router: 32 threads per expert
  {
    int e = t >> 5, j0 = (t & 31) * 32;
    const float* wr = gw + (size_t)e * DIM + j0;
    float p = 0.f;
#pragma unroll 8
    for (int j = 0; j < 32; ++j) p += norm[j0 + j] * wr[j];
    red[t] = p;
  }
  __syncthreads();
  for (int st = 16; st > 0; st >>= 1) { if ((t & 31) < st) red[t] += red[t + st]; __syncthreads(); }
  if ((t & 31) == 0) lg[t >> 5] = red[t] + gb[t >> 5];
  __syncthreads();
  if (t < NEXP) logits[(size_t)row * NEXP + t] = lg[t];
  if (t == 0) {
    int i0 = 0;
    for (int e = 1; e < NEXP; ++e) if (lg[e] > lg[i0]) i0 = e;
    int i1 = (i0 == 0) ? 1 : 0;
    for (int e = 0; e < NEXP; ++e) if (e != i0 && lg[e] > lg[i1]) i1 = e;
    float e1 = __expf(lg[i1] - lg[i0]);
    float den = 1.0f + e1;
    float* cwr = cw + (size_t)row * NEXP;
    for (int e = 0; e < NEXP; ++e) cwr[e] = 0.f;
    cwr[i0] = 1.0f / den;
    cwr[i1] = e1 / den;
  }
}

// expert FFN stage 1: hid = gelu(h2 @ w1[e] + b1[e])   (w1[e] is [K=DIM, N=DFF])
__global__ void __launch_bounds__(256) k_moe1(const __bf16* __restrict__ A,
                                              const float* __restrict__ W,
                                              const float* __restrict__ bias,
                                              __bf16* __restrict__ hid) {
  __shared__ __bf16 As[2 * TILE_ELEMS], Bs[2 * TILE_ELEMS];
  const int n0 = blockIdx.x * 128, m0 = blockIdx.y * 128;
  v8f acc[8];
  gemm_main<false>(A, W, DFF, DIM, m0, n0, As, Bs, acc);
  EPI_BEGIN
    hid[(size_t)m * DFF + n] = f2bf(gelu_tanh(v + bias[n]));
  EPI_END
}

// expert FFN stage 2: out += cw[:,e] * (hid @ w2[e] + b2[e])   (w2[e] is [K=DFF, N=DIM])
__global__ void __launch_bounds__(256) k_moe2(const __bf16* __restrict__ A,
                                              const float* __restrict__ W,
                                              const float* __restrict__ bias,
                                              const float* __restrict__ cw,
                                              int expert,
                                              float* __restrict__ out) {
  __shared__ __bf16 As[2 * TILE_ELEMS], Bs[2 * TILE_ELEMS];
  const int n0 = blockIdx.x * 128, m0 = blockIdx.y * 128;
  v8f acc[8];
  gemm_main<false>(A, W, DIM, DFF, m0, n0, As, Bs, acc);
  EPI_BEGIN
    float w = cw[(size_t)m * NEXP + expert];
    size_t idx = (size_t)m * DIM + n;
    out[idx] = out[idx] + w * (v + bias[n]);
  EPI_END
}

// ---------------------------------------------------------------------------
extern "C" void kernel_launch(void* const* d_in, const int* in_sizes, int n_in,
                              void* d_out, int out_size, void* d_ws, size_t ws_size,
                              hipStream_t stream) {
  (void)in_sizes; (void)n_in; (void)out_size; (void)ws_size;
  const float* x      = (const float*)d_in[0];
  // d_in[1] = pos_indices (unused by the reference)
  const float* ln1_g  = (const float*)d_in[2];
  const float* ln1_b  = (const float*)d_in[3];
  const float* in_w   = (const float*)d_in[4];
  const float* in_b   = (const float*)d_in[5];
  const float* out_w  = (const float*)d_in[6];
  const float* out_b  = (const float*)d_in[7];
  const float* ln2_g  = (const float*)d_in[8];
  const float* ln2_b  = (const float*)d_in[9];
  const float* gate_w = (const float*)d_in[10];
  const float* gate_b = (const float*)d_in[11];
  const float* w1     = (const float*)d_in[12];
  const float* b1     = (const float*)d_in[13];
  const float* w2     = (const float*)d_in[14];
  const float* b2     = (const float*)d_in[15];

  char* ws = (char*)d_ws;
  __bf16* h1  = (__bf16*)(ws + (size_t)(0)  * (1 << 20));   // S*D bf16     (4MB)
  __bf16* Qb  = (__bf16*)(ws + (size_t)(4)  * (1 << 20));   // H*S*64 bf16  (4MB)
  __bf16* Kb  = (__bf16*)(ws + (size_t)(8)  * (1 << 20));   // H*S*64 bf16  (4MB)
  __bf16* Vt  = (__bf16*)(ws + (size_t)(12) * (1 << 20));   // H*64*S bf16  (4MB)
  __bf16* ctx = (__bf16*)(ws + (size_t)(16) * (1 << 20));   // S*D bf16     (4MB)
  __bf16* h2  = (__bf16*)(ws + (size_t)(20) * (1 << 20));   // S*D bf16     (4MB)
  __bf16* hid = (__bf16*)(ws + (size_t)(24) * (1 << 20));   // S*DFF bf16   (8MB)
  float*  cw  = (float*) (ws + (size_t)(32) * (1 << 20));   // S*8 f32

  float* xout   = (float*)d_out;                 // [S, DIM]
  float* logits = xout + (size_t)S_LEN * DIM;    // [S, NEXP]

  k_ln1<<<S_LEN, 256, 0, stream>>>(x, ln1_g, ln1_b, h1);
  k_gemm_qkv<<<dim3(3 * DIM / 128, S_LEN / 128), 256, 0, stream>>>(h1, in_w, in_b, Qb, Kb, Vt);
  k_attn<<<dim3(S_LEN / 64, HEADS), 128, 0, stream>>>(Qb, Kb, Vt, ctx);
  k_gemm_out<<<dim3(DIM / 128, S_LEN / 128), 256, 0, stream>>>(ctx, out_w, out_b, x, xout);
  k_ln2_router<<<S_LEN, 256, 0, stream>>>(xout, ln2_g, ln2_b, gate_w, gate_b, h2, logits, cw);
  for (int e = 0; e < NEXP; ++e) {
    k_moe1<<<dim3(DFF / 128, S_LEN / 128), 256, 0, stream>>>(
        h2, w1 + (size_t)e * DIM * DFF, b1 + (size_t)e * DFF, hid);
    k_moe2<<<dim3(DIM / 128, S_LEN / 128), 256, 0, stream>>>(
        hid, w2 + (size_t)e * DFF * DIM, b2 + (size_t)e * DIM, cw, e, xout);
  }
}
